// CausalSelfAttention_9689446220332
// MI455X (gfx1250) — compile-verified, matrix operand order checked by reading, not run on hardware
//
#include <hip/hip_runtime.h>
#include <hip/hip_bf16.h>

typedef __bf16 bf16_t;
typedef __attribute__((ext_vector_type(16))) __bf16 v16bf;
typedef __attribute__((ext_vector_type(8)))  __bf16 v8bf;
typedef __attribute__((ext_vector_type(8)))  float  v8f;

#define LOG2E 1.4426950408889634f
#define NEG_BIG (-1.0e30f)

static __device__ inline v8f zero8() {
  v8f z;
#pragma unroll
  for (int i = 0; i < 8; ++i) z[i] = 0.0f;
  return z;
}

static __device__ inline v8f wmma_bf16(v16bf a, v16bf b, v8f c) {
  // (neg_a, A, neg_b, B, c_mod, C, reuse_a, reuse_b)
  return __builtin_amdgcn_wmma_f32_16x16x32_bf16(false, a, false, b, (short)0, c,
                                                 false, false);
}

// ---------------------------------------------------------------------------
// Fragment layout (16-bit A/B, wave32): lane holds row (lid&15); element e
// holds K = (e&7) + 16*(e>>3) + 8*half -> two contiguous 16-byte runs at
// rowbase + 8*half and rowbase + 8*half + 16.
// All loaders take a pointer already offset to (rowbase + 8*half).
// ---------------------------------------------------------------------------
static __device__ inline v16bf load_frag_bf16(const bf16_t* __restrict__ p) {
  v8bf lo = *(const v8bf*)(p);
  v8bf hi = *(const v8bf*)(p + 16);
  v16bf o;
#pragma unroll
  for (int i = 0; i < 8; ++i) { o[i] = lo[i]; o[8 + i] = hi[i]; }
  return o;
}

// ---------------------------------------------------------------------------
// Kernel 0: one-shot fp32 -> bf16 conversion (8 elements / thread, b128 IO).
// Removes all v_cvt from the GEMM inner loops (weights were being re-
// converted up to 512x otherwise).
// ---------------------------------------------------------------------------
__global__ __launch_bounds__(256) void cvt_bf16_kernel(
    const float* __restrict__ src, bf16_t* __restrict__ dst, int n8) {
  const int i = blockIdx.x * blockDim.x + threadIdx.x;
  if (i >= n8) return;
  const float4* s = (const float4*)src + (size_t)2 * i;
  const float4 f0 = s[0];
  const float4 f1 = s[1];
  v8bf o;
  o[0] = (__bf16)f0.x; o[1] = (__bf16)f0.y; o[2] = (__bf16)f0.z; o[3] = (__bf16)f0.w;
  o[4] = (__bf16)f1.x; o[5] = (__bf16)f1.y; o[6] = (__bf16)f1.z; o[7] = (__bf16)f1.w;
  *((v8bf*)dst + i) = o;
}

// ---------------------------------------------------------------------------
// Kernel 1: fused QKV projection.  z = blockIdx.z selects {Q, K, V}.
// One 32x64 tile per wave (two A fragments share each B fragment);
// K-loop software-pipelined one step ahead; pure bf16 load + WMMA inner loop.
// Q/K stored [b,h,t,d] bf16, V stored transposed [b,h,d,t] bf16.
// ---------------------------------------------------------------------------
__global__ __launch_bounds__(128) void qkv_proj_kernel(
    const bf16_t* __restrict__ xbf,
    const bf16_t* __restrict__ Wqb, const float* __restrict__ bq,
    const bf16_t* __restrict__ Wkb, const float* __restrict__ bk,
    const bf16_t* __restrict__ Wvb, const float* __restrict__ bv,
    bf16_t* __restrict__ qws, bf16_t* __restrict__ kws, bf16_t* __restrict__ vTws) {
  const int lid   = threadIdx.x & 31;
  const int wv    = threadIdx.x >> 5;
  const int row16 = lid & 15;
  const int half  = lid >> 4;

  const int m0 = (blockIdx.x * 4 + wv) * 32;  // rows of x, [0, 8192)
  const int n0 = blockIdx.y * 64;             // out channels, [0, 512)
  const int z  = blockIdx.z;

  const bf16_t* W    = (z == 0) ? Wqb : (z == 1) ? Wkb : Wvb;
  const float*  bias = (z == 0) ? bq : (z == 1) ? bk : bv;

  // Stream base pointers (per-lane row base + 8*half); bumped by 32 per step.
  const bf16_t* xp = xbf + (size_t)(m0 + row16) * 512 + 8 * half;
  const bf16_t* wp = W   + (size_t)(n0 + row16) * 512 + 8 * half;

  v8f acc[2][4];
#pragma unroll
  for (int mt = 0; mt < 2; ++mt)
#pragma unroll
    for (int g = 0; g < 4; ++g) acc[mt][g] = zero8();

  // Software pipeline: loads for step kk+32 issued during step kk's WMMAs.
  v16bf a0 = load_frag_bf16(xp);
  v16bf a1 = load_frag_bf16(xp + 16 * 512);
  v16bf bfr[4];
#pragma unroll
  for (int g = 0; g < 4; ++g) bfr[g] = load_frag_bf16(wp + (size_t)(16 * g) * 512);

#pragma unroll 4
  for (int kk = 0; kk < 512; kk += 32) {
    const v16bf ca0 = a0, ca1 = a1;
    const v16bf cb0 = bfr[0], cb1 = bfr[1], cb2 = bfr[2], cb3 = bfr[3];
    if (kk + 32 < 512) {
      xp += 32; wp += 32;
      a0 = load_frag_bf16(xp);
      a1 = load_frag_bf16(xp + 16 * 512);
#pragma unroll
      for (int g = 0; g < 4; ++g) bfr[g] = load_frag_bf16(wp + (size_t)(16 * g) * 512);
    }
    acc[0][0] = wmma_bf16(ca0, cb0, acc[0][0]);
    acc[0][1] = wmma_bf16(ca0, cb1, acc[0][1]);
    acc[0][2] = wmma_bf16(ca0, cb2, acc[0][2]);
    acc[0][3] = wmma_bf16(ca0, cb3, acc[0][3]);
    acc[1][0] = wmma_bf16(ca1, cb0, acc[1][0]);
    acc[1][1] = wmma_bf16(ca1, cb1, acc[1][1]);
    acc[1][2] = wmma_bf16(ca1, cb2, acc[1][2]);
    acc[1][3] = wmma_bf16(ca1, cb3, acc[1][3]);
  }

#pragma unroll
  for (int g = 0; g < 4; ++g) {
    const int   n  = n0 + 16 * g + row16;
    const float bb = bias[n];
    const int   hh = n >> 6;
    const int   dc = n & 63;
#pragma unroll
    for (int mt = 0; mt < 2; ++mt) {
#pragma unroll
      for (int r = 0; r < 8; ++r) {
        const int m  = m0 + 16 * mt + 8 * half + r;
        const int bi = m >> 10;
        const int tt = m & 1023;
        const float val = acc[mt][g][r] + bb;
        if (z == 2) {
          vTws[(((size_t)bi * 8 + hh) * 64 + dc) * 1024 + tt] = (bf16_t)val;
        } else {
          bf16_t* dst = (z == 0) ? qws : kws;
          dst[(((size_t)bi * 8 + hh) * 1024 + tt) * 64 + dc] = (bf16_t)val;
        }
      }
    }
  }
}

// ---------------------------------------------------------------------------
// Kernel 2: flash attention with frame-granularity (8) block-causal mask.
// One wave per 16-query tile per (b,h); keys processed 32 at a time.
// ---------------------------------------------------------------------------
__global__ __launch_bounds__(128) void attn_kernel(
    const bf16_t* __restrict__ qws, const bf16_t* __restrict__ kws,
    const bf16_t* __restrict__ vTws, const float* __restrict__ mask,
    bf16_t* __restrict__ yws) {
  __shared__ bf16_t ldsP[4][16 * 32];  // per-wave 16x32 bf16 P tile

  const int lid   = threadIdx.x & 31;
  const int wv    = threadIdx.x >> 5;
  const int row16 = lid & 15;
  const int half  = lid >> 4;

  const int bh = blockIdx.y;  // b*8 + h
  const int b  = bh >> 3;
  const int h  = bh & 7;
  const int q0 = (blockIdx.x * 4 + wv) * 16;

  const bf16_t* Qb = qws  + (size_t)bh * 1024 * 64;
  const bf16_t* Kb = kws  + (size_t)bh * 1024 * 64;
  const bf16_t* Vb = vTws + (size_t)bh * 64 * 1024;
  const float*  mk = mask + (size_t)b * 1024;

  // Q fragments held for the whole key loop (d = 64 -> two K=32 fragments).
  const bf16_t* qp = Qb + (size_t)(q0 + row16) * 64 + 8 * half;
  const v16bf aq0 = load_frag_bf16(qp);
  const v16bf aq1 = load_frag_bf16(qp + 32);

  // Per-lane stream bases, bumped per 32-key step.
  const bf16_t* kp = Kb + (size_t)row16 * 64 + 8 * half;      // += 32*64
  const bf16_t* vp = Vb + (size_t)row16 * 1024 + 8 * half;    // += 32

  v8f o[4];
#pragma unroll
  for (int g = 0; g < 4; ++g) o[g] = zero8();
  float mrow[8], lrow[8];
#pragma unroll
  for (int r = 0; r < 8; ++r) { mrow[r] = NEG_BIG; lrow[r] = 0.0f; }

  const int kend = q0 + 16;
  for (int kb = 0; kb < kend; kb += 32, kp += 32 * 64, vp += 32) {
    const bool tileb = (kb + 16) < kend;  // wave-uniform

    // Issue all K-fragment loads up front (independent b128 pairs).
    const v16bf bk0a = load_frag_bf16(kp);
    const v16bf bk1a = load_frag_bf16(kp + 32);
    v16bf bk0b, bk1b;
    if (tileb) {
      bk0b = load_frag_bf16(kp + 16 * 64);
      bk1b = load_frag_bf16(kp + 16 * 64 + 32);
    }
    const float mva = mk[kb + row16];
    const float mvb = mk[kb + 16 + row16];

    // S = Q K^T for 32 keys (two 16-key D tiles).
    v8f c0 = zero8(), c1 = zero8();
    c0 = wmma_bf16(aq0, bk0a, c0);
    c0 = wmma_bf16(aq1, bk1a, c0);
    if (tileb) {
      c1 = wmma_bf16(aq0, bk0b, c1);
      c1 = wmma_bf16(aq1, bk1b, c1);
    }

    const int fka = (kb + row16) >> 3;       // key frame, tile a
    const int fkb = (kb + 16 + row16) >> 3;  // key frame, tile b

    float s0[8], s1[8];
#pragma unroll
    for (int r = 0; r < 8; ++r) {
      const int fq = (q0 + 8 * half + r) >> 3;  // query frame
      float sa = c0[r] * 0.125f;                // 1/sqrt(64)
      float sb = c1[r] * 0.125f;
      if (fka > fq || mva == 0.0f)           sa = NEG_BIG;
      if (!tileb || fkb > fq || mvb == 0.0f) sb = NEG_BIG;
      s0[r] = sa; s1[r] = sb;
    }

    // Online softmax: row stats via width-16 xor reductions (rows live in
    // VGPR index, cols in lanes; each 16-lane half owns 8 rows).
    float p0[8], p1[8];
#pragma unroll
    for (int r = 0; r < 8; ++r) {
      float mx = fmaxf(s0[r], s1[r]);
#pragma unroll
      for (int off = 8; off > 0; off >>= 1) mx = fmaxf(mx, __shfl_xor(mx, off, 16));
      const float mnew = fmaxf(mrow[r], mx);
      const float pa = (s0[r] <= NEG_BIG * 0.5f) ? 0.0f : exp2f((s0[r] - mnew) * LOG2E);
      const float pb = (s1[r] <= NEG_BIG * 0.5f) ? 0.0f : exp2f((s1[r] - mnew) * LOG2E);
      float ps = pa + pb;
#pragma unroll
      for (int off = 8; off > 0; off >>= 1) ps += __shfl_xor(ps, off, 16);
      const float corr = exp2f((mrow[r] - mnew) * LOG2E);
      lrow[r] = lrow[r] * corr + ps;
      mrow[r] = mnew;
#pragma unroll
      for (int g = 0; g < 4; ++g) o[g][r] *= corr;
      p0[r] = pa; p1[r] = pb;
    }

    // D-tile -> A-fragment relayout of P through per-wave LDS.
    bf16_t* lp = ldsP[wv];
#pragma unroll
    for (int r = 0; r < 8; ++r) {
      const int rr = 8 * half + r;
      lp[rr * 32 + row16]      = (bf16_t)p0[r];
      lp[rr * 32 + 16 + row16] = (bf16_t)p1[r];
    }
    const v16bf ap = load_frag_bf16(lp + row16 * 32 + 8 * half);

    // O += P V  (V^T layout makes B-fragment loads contiguous along keys).
#pragma unroll
    for (int g = 0; g < 4; ++g) {
      const v16bf bvf = load_frag_bf16(vp + (size_t)(16 * g) * 1024);
      o[g] = wmma_bf16(ap, bvf, o[g]);
    }
  }

  // Normalize and emit y[b, t, h*64 + d] as bf16 for the output projection.
#pragma unroll
  for (int r = 0; r < 8; ++r) {
    const int   t   = q0 + 8 * half + r;
    const float inv = 1.0f / lrow[r];
#pragma unroll
    for (int g = 0; g < 4; ++g) {
      yws[((size_t)b * 1024 + t) * 512 + h * 64 + 16 * g + row16] =
          (bf16_t)(o[g][r] * inv);
    }
  }
}

// ---------------------------------------------------------------------------
// Kernel 3: output projection  out = y @ Wp^T + bp  (fp32 result),
// 32x64 tile per wave, software-pipelined, pure bf16 inner loop.
// ---------------------------------------------------------------------------
__global__ __launch_bounds__(128) void out_proj_kernel(
    const bf16_t* __restrict__ yws, const bf16_t* __restrict__ Wpb,
    const float* __restrict__ bp, float* __restrict__ out) {
  const int lid   = threadIdx.x & 31;
  const int wv    = threadIdx.x >> 5;
  const int row16 = lid & 15;
  const int half  = lid >> 4;

  const int m0 = (blockIdx.x * 4 + wv) * 32;
  const int n0 = blockIdx.y * 64;

  const bf16_t* yp = yws + (size_t)(m0 + row16) * 512 + 8 * half;
  const bf16_t* wp = Wpb + (size_t)(n0 + row16) * 512 + 8 * half;

  v8f acc[2][4];
#pragma unroll
  for (int mt = 0; mt < 2; ++mt)
#pragma unroll
    for (int g = 0; g < 4; ++g) acc[mt][g] = zero8();

  v16bf a0 = load_frag_bf16(yp);
  v16bf a1 = load_frag_bf16(yp + 16 * 512);
  v16bf bfr[4];
#pragma unroll
  for (int g = 0; g < 4; ++g) bfr[g] = load_frag_bf16(wp + (size_t)(16 * g) * 512);

#pragma unroll 4
  for (int kk = 0; kk < 512; kk += 32) {
    const v16bf ca0 = a0, ca1 = a1;
    const v16bf cb0 = bfr[0], cb1 = bfr[1], cb2 = bfr[2], cb3 = bfr[3];
    if (kk + 32 < 512) {
      yp += 32; wp += 32;
      a0 = load_frag_bf16(yp);
      a1 = load_frag_bf16(yp + 16 * 512);
#pragma unroll
      for (int g = 0; g < 4; ++g) bfr[g] = load_frag_bf16(wp + (size_t)(16 * g) * 512);
    }
    acc[0][0] = wmma_bf16(ca0, cb0, acc[0][0]);
    acc[0][1] = wmma_bf16(ca0, cb1, acc[0][1]);
    acc[0][2] = wmma_bf16(ca0, cb2, acc[0][2]);
    acc[0][3] = wmma_bf16(ca0, cb3, acc[0][3]);
    acc[1][0] = wmma_bf16(ca1, cb0, acc[1][0]);
    acc[1][1] = wmma_bf16(ca1, cb1, acc[1][1]);
    acc[1][2] = wmma_bf16(ca1, cb2, acc[1][2]);
    acc[1][3] = wmma_bf16(ca1, cb3, acc[1][3]);
  }

#pragma unroll
  for (int g = 0; g < 4; ++g) {
    const int   n  = n0 + 16 * g + row16;
    const float bb = bp[n];
#pragma unroll
    for (int mt = 0; mt < 2; ++mt) {
#pragma unroll
      for (int r = 0; r < 8; ++r) {
        const int m = m0 + 16 * mt + 8 * half + r;
        out[(size_t)m * 512 + n] = acc[mt][g][r] + bb;
      }
    }
  }
}

extern "C" void kernel_launch(void* const* d_in, const int* in_sizes, int n_in,
                              void* d_out, int out_size, void* d_ws, size_t ws_size,
                              hipStream_t stream) {
  const float* x    = (const float*)d_in[0];
  const float* mask = (const float*)d_in[1];
  const float* Wq   = (const float*)d_in[2];
  const float* bq   = (const float*)d_in[3];
  const float* Wk   = (const float*)d_in[4];
  const float* bk   = (const float*)d_in[5];
  const float* Wv   = (const float*)d_in[6];
  const float* bv   = (const float*)d_in[7];
  const float* Wp   = (const float*)d_in[8];
  const float* bp   = (const float*)d_in[9];

  const size_t qkv_elems = (size_t)8 * 8 * 1024 * 64;  // 4 Mi bf16 each
  const size_t x_elems   = (size_t)8 * 1024 * 512;     // 4 Mi
  const size_t w_elems   = (size_t)512 * 512;          // 256 Ki

  bf16_t* qws  = (bf16_t*)d_ws;
  bf16_t* kws  = qws + qkv_elems;
  bf16_t* vTws = kws + qkv_elems;
  bf16_t* yws  = vTws + qkv_elems;
  bf16_t* xbf  = yws + x_elems;   // y is also B*T*C
  bf16_t* wqb  = xbf + x_elems;
  bf16_t* wkb  = wqb + w_elems;
  bf16_t* wvb  = wkb + w_elems;
  bf16_t* wpb  = wvb + w_elems;
  (void)in_sizes; (void)n_in; (void)out_size; (void)ws_size;

  // 0) One-shot fp32 -> bf16 conversions (x + 4 weight matrices).
  {
    const int n8x = (int)(x_elems / 8);
    cvt_bf16_kernel<<<(n8x + 255) / 256, 256, 0, stream>>>(x, xbf, n8x);
    const int n8w = (int)(w_elems / 8);
    cvt_bf16_kernel<<<(n8w + 255) / 256, 256, 0, stream>>>(Wq, wqb, n8w);
    cvt_bf16_kernel<<<(n8w + 255) / 256, 256, 0, stream>>>(Wk, wkb, n8w);
    cvt_bf16_kernel<<<(n8w + 255) / 256, 256, 0, stream>>>(Wv, wvb, n8w);
    cvt_bf16_kernel<<<(n8w + 255) / 256, 256, 0, stream>>>(Wp, wpb, n8w);
  }

  // 1) QKV projections (z: 0=Q, 1=K, 2=V^T), 32x64 tile per wave.
  dim3 g1(8192 / 32 / 4, 512 / 64, 3);
  qkv_proj_kernel<<<g1, 128, 0, stream>>>(xbf, wqb, bq, wkb, bk, wvb, bv,
                                          qws, kws, vTws);

  // 2) Flash attention: one wave per 16-query tile per (b,h).
  dim3 g2(1024 / 16 / 4, 64);
  attn_kernel<<<g2, 128, 0, stream>>>(qws, kws, vTws, mask, yws);

  // 3) Output projection to fp32.
  dim3 g3(8192 / 32 / 4, 512 / 64);
  out_proj_kernel<<<g3, 128, 0, stream>>>(yws, wpb, bp, (float*)d_out);
}
